// TimeSeriesEmbedding_14946486190105
// MI455X (gfx1250) — 8-bit matrix operand order repaired and checked by reading, compiled, not run on hardware
//
#include <hip/hip_runtime.h>
#include <stdint.h>


typedef int v8i __attribute__((ext_vector_type(8)));

#define BATCH 64
#define SEQ   1024
#define KDIM  32
#define ODIM  1024
#define NTOK  (BATCH * SEQ)

static constexpr size_t WQ_OFF  = 0;
static constexpr size_t ISW_OFF = 32768;
static constexpr size_t ISX_OFF = 65536;
static constexpr size_t XQ_OFF  = 327680;
static constexpr size_t PE_OFF  = 2424832;

__global__ __launch_bounds__(1024) void w_quant(const float* __restrict__ W,
                                                int8_t* __restrict__ wq,
                                                float* __restrict__ inv_sw) {
  __shared__ float red[1024];
  const int t = threadIdx.x;
  float s = 0.f;
#pragma unroll
  for (int i = 0; i < 32; ++i) s += fabsf(W[t + i * 1024]);
  red[t] = s;
  __syncthreads();
  for (int off = 512; off > 0; off >>= 1) {
    if (t < off) red[t] += red[t + off];
    __syncthreads();
  }
  const float m = fmaxf(red[0] * (1.0f / 32768.0f), 1e-5f);
  if (t == 0) *inv_sw = m;
  const float scale = 1.0f / m;

  const float* wrow = W + (size_t)t * KDIM;
  int packed[8];
#pragma unroll
  for (int d = 0; d < 8; ++d) {
    int pk = 0;
#pragma unroll
    for (int bb = 0; bb < 4; ++bb) {
      int q = (int)rintf(wrow[d * 4 + bb] * scale);
      q = q < -1 ? -1 : (q > 1 ? 1 : q);
      pk |= (q & 0xff) << (8 * bb);
    }
    packed[d] = pk;
  }
  int4* dst = (int4*)(wq + (size_t)t * KDIM);
  dst[0] = make_int4(packed[0], packed[1], packed[2], packed[3]);
  dst[1] = make_int4(packed[4], packed[5], packed[6], packed[7]);
}

__global__ __launch_bounds__(256) void act_quant(const float* __restrict__ x,
                                                 int8_t* __restrict__ xq,
                                                 float* __restrict__ inv_sx) {
  __shared__ float tile[8][32][33];
  const int lane = threadIdx.x & 31;
  const int wave = threadIdx.x >> 5;
  const int tokBase = blockIdx.x * 256 + wave * 32;

#pragma unroll 4
  for (int r = 0; r < 32; ++r)
    tile[wave][r][lane] = x[(size_t)(tokBase + r) * KDIM + lane];
  __syncthreads();

  const int tok = tokBase + lane;
  float amax = 0.f;
#pragma unroll
  for (int k = 0; k < 32; ++k) amax = fmaxf(amax, fabsf(tile[wave][lane][k]));
  amax = fmaxf(amax, 1e-5f);
  const float scale = 127.0f / amax;
  inv_sx[tok] = amax * (1.0f / 127.0f);

  int packed[8];
#pragma unroll
  for (int d = 0; d < 8; ++d) {
    int pk = 0;
#pragma unroll
    for (int bb = 0; bb < 4; ++bb) {
      int q = (int)rintf(tile[wave][lane][d * 4 + bb] * scale);
      q = q < -128 ? -128 : (q > 127 ? 127 : q);
      pk |= (q & 0xff) << (8 * bb);
    }
    packed[d] = pk;
  }
  int4* dst = (int4*)(xq + (size_t)tok * KDIM);
  dst[0] = make_int4(packed[0], packed[1], packed[2], packed[3]);
  dst[1] = make_int4(packed[4], packed[5], packed[6], packed[7]);
}

__global__ __launch_bounds__(256) void peb_kernel(const float* __restrict__ bias,
                                                  float* __restrict__ peb) {
  const int idx = blockIdx.x * 256 + threadIdx.x;
  const int s = idx >> 10;
  const int i = idx & 1023;
  const float even = (float)(i & ~1);
  const float c = 13.28771237954945f / 1024.0f;
  const float ang = (float)s * exp2f(-even * c);
  peb[idx] = ((i & 1) ? cosf(ang) : sinf(ang)) + bias[i];
}

__global__ __launch_bounds__(256) void bitnet_gemm(
    const int8_t* __restrict__ xq, const float* __restrict__ inv_sx,
    const int8_t* __restrict__ wq, const float* __restrict__ inv_sw_p,
    const float* __restrict__ peb, float* __restrict__ out) {
  const int lane = threadIdx.x & 31;
  const int wave = threadIdx.x >> 5;
  const int lo = lane & 15;
  const int hi = lane >> 4;
  const int tokBase = blockIdx.x * 16;
  const int rowBase = tokBase + hi * 8;
  const int sBase   = (tokBase & (SEQ - 1)) + hi * 8;
  const float invsw = *inv_sw_p;

  const int8_t* arow = xq + (size_t)(tokBase + lo) * KDIM + hi * 16;
  v8i a;
  {
    int2 p0 = *(const int2*)(arow);
    int2 p1 = *(const int2*)(arow + 8);
    a[0] = p0.x; a[1] = p0.y; a[2] = p1.x; a[3] = p1.y;
    a[4] = 0; a[5] = 0; a[6] = 0; a[7] = 0;
  }

  float sxv[8];
#pragma unroll
  for (int j = 0; j < 8; ++j)
    sxv[j] = inv_sx[rowBase + j] * invsw;

  const float*  pep = peb + (size_t)sBase * ODIM + wave * 128 + lo;
  float*        op  = out + (size_t)rowBase * ODIM + wave * 128 + lo;
  const int8_t* wp  = wq + (size_t)(wave * 128 + lo) * KDIM + hi * 16;

#pragma unroll
  for (int i = 0; i < 8; ++i) {
    v8i b;
    {
      int4 q = *(const int4*)(wp + i * 16 * KDIM);
      b[0] = q.x; b[1] = q.y; b[2] = q.z; b[3] = q.w;
      b[4] = 0; b[5] = 0; b[6] = 0; b[7] = 0;
    }

    float pv[8];
#pragma unroll
    for (int j = 0; j < 8; ++j)
      pv[j] = pep[(size_t)j * ODIM + i * 16];

    v8i c = {};
    c = __builtin_amdgcn_wmma_i32_16x16x64_iu8(
         true, a,  true, b, c,
         false,  false);

#pragma unroll
    for (int j = 0; j < 8; ++j)
      op[(size_t)j * ODIM + i * 16] =
          fmaf((float)c[j], sxv[j], pv[j]);
  }
}

extern "C" void kernel_launch(void* const* d_in, const int* in_sizes, int n_in,
                              void* d_out, int out_size, void* d_ws, size_t ws_size,
                              hipStream_t stream) {
  const float* x = (const float*)d_in[0];
  const float* W = (const float*)d_in[1];
  const float* b = (const float*)d_in[2];
  float* out = (float*)d_out;

  char* ws = (char*)d_ws;
  int8_t* wq     = (int8_t*)(ws + WQ_OFF);
  float*  inv_sw = (float*)(ws + ISW_OFF);
  float*  inv_sx = (float*)(ws + ISX_OFF);
  int8_t* xq     = (int8_t*)(ws + XQ_OFF);
  float*  peb    = (float*)(ws + PE_OFF);

  hipLaunchKernelGGL(w_quant,     dim3(1),                dim3(1024), 0, stream, W, wq, inv_sw);
  hipLaunchKernelGGL(act_quant,   dim3(NTOK / 256),       dim3(256),  0, stream, x, xq, inv_sx);
  hipLaunchKernelGGL(peb_kernel,  dim3(SEQ * ODIM / 256), dim3(256),  0, stream, b, peb);
  hipLaunchKernelGGL(bitnet_gemm, dim3(NTOK / 16),        dim3(256),  0, stream,
                     xq, inv_sx, wq, inv_sw, peb, out);
}
